// BrainSphereCNN_50714973831424
// MI455X (gfx1250) — matrix operand order repaired and checked by reading, not compiled, not run on hardware
//
#include <hip/hip_runtime.h>

// ---------------- types ----------------
typedef __attribute__((ext_vector_type(16))) __bf16 v16bf;
typedef __attribute__((ext_vector_type(8)))  __bf16 v8bf;
typedef __attribute__((ext_vector_type(8)))  float  v8f;

#define NN 40962                 // vertices
#define MT ((NN + 15) / 16)      // 2561 M-tiles of 16 rows
#define GBLK ((MT + 7) / 8)      // 8 waves (M-tiles) per 256-thread block

__device__ __forceinline__ unsigned short f2bf(float f) {
    unsigned int u = __float_as_uint(f);
    unsigned int r = u + 0x7FFFu + ((u >> 16) & 1u);   // RNE
    return (unsigned short)(r >> 16);
}

__device__ __forceinline__ v16bf mk16(const unsigned short* p0, const unsigned short* p1) {
    v8bf a0 = *(const v8bf*)(p0);
    v8bf a1 = *(const v8bf*)(p1);
    v16bf a;
#pragma unroll
    for (int i = 0; i < 8; ++i) { a[i] = a0[i]; a[i + 8] = a1[i]; }
    return a;
}

// ---------------- weight / input prep ----------------
// W0 [21,128] -> W0t bf16 [128,32] (K padded 21->32 with zeros), n-major
__global__ void k_w0t(const float* __restrict__ W0, unsigned short* __restrict__ W0t) {
    int i = blockIdx.x * blockDim.x + threadIdx.x;
    if (i >= 128 * 32) return;
    int n = i >> 5, k = i & 31;
    W0t[i] = (k < 21) ? f2bf(W0[k * 128 + n]) : (unsigned short)0;
}

// Wm [14,896,128] -> Wmt bf16 [14,128,896]  (per-layer n-major)
__global__ void k_wmt(const float* __restrict__ Wm, unsigned short* __restrict__ Wmt) {
    int i = blockIdx.x * blockDim.x + threadIdx.x;
    if (i >= 14 * 128 * 896) return;
    int l = i / (128 * 896);
    int r = i - l * (128 * 896);
    int n = r / 896, k = r - n * 896;
    Wmt[i] = f2bf(Wm[(size_t)l * 896 * 128 + k * 128 + n]);
}

// Wl [896,36] -> Wlt bf16 [48,896] (N padded 36->48 with zeros), n-major
__global__ void k_wlt(const float* __restrict__ Wl, unsigned short* __restrict__ Wlt) {
    int i = blockIdx.x * blockDim.x + threadIdx.x;
    if (i >= 48 * 896) return;
    int n = i / 896, k = i - n * 896;
    Wlt[i] = (n < 36) ? f2bf(Wl[k * 36 + n]) : (unsigned short)0;
}

// Layer-0 gathered input: A0 bf16 [NN,32]; k<21: x[idx[v*7 + k/3]*3 + k%3], else 0
__global__ void k_a0(const float* __restrict__ x, const int* __restrict__ idx,
                     unsigned short* __restrict__ A0) {
    int i = blockIdx.x * blockDim.x + threadIdx.x;
    if (i >= NN * 32) return;
    int v = i >> 5, k = i & 31;
    unsigned short r = 0;
    if (k < 21) {
        int j = k / 3, c = k - j * 3;
        r = f2bf(x[(size_t)idx[v * 7 + j] * 3 + c]);
    }
    A0[i] = r;
}

// ---------------- WMMA GEMMs ----------------
// Layer 0: pre[NN,128] = A0[NN,32] @ W0 (bf16 in, f32 out). Single K-chunk.
__global__ __launch_bounds__(256) void gemm0(const unsigned short* __restrict__ A0,
                                             const unsigned short* __restrict__ W0t,
                                             float* __restrict__ pre) {
    const int lane = threadIdx.x & 31;
    const int wave = threadIdx.x >> 5;
    const int tile = blockIdx.x * 8 + wave;
    if (tile >= MT) return;                      // uniform per wave
    const int row = lane & 15;
    const int hi  = lane >> 4;
    const int m   = tile * 16 + row;
    const int mc  = (m < NN) ? m : (NN - 1);
    const int alo = hi ? 8 : 0;
    const int blo = hi ? 8 : 0;

    const v8f zacc = {0.f, 0.f, 0.f, 0.f, 0.f, 0.f, 0.f, 0.f};
    v8f acc[8];
#pragma unroll
    for (int t = 0; t < 8; ++t) acc[t] = zacc;

    const unsigned short* ap = A0 + mc * 32 + alo;
    v16bf a = mk16(ap, ap + 16);

#pragma unroll
    for (int t = 0; t < 8; ++t) {
        const unsigned short* bp = W0t + (t * 16 + row) * 32 + blo;
        v16bf b = mk16(bp, bp + 16);
        acc[t] = __builtin_amdgcn_wmma_f32_16x16x32_bf16(false, a, false, b,
                                                         (short)0, acc[t], false, false);
    }

    if (tile * 16 + 16 <= NN) {                  // uniform: full tile, unguarded stores
        float* pbase = pre + (size_t)(tile * 16 + hi * 8) * 128 + row;
#pragma unroll
        for (int t = 0; t < 8; ++t)
#pragma unroll
            for (int r = 0; r < 8; ++r)
                pbase[(size_t)r * 128 + t * 16] = acc[t][r];
    } else {
#pragma unroll
        for (int t = 0; t < 8; ++t)
#pragma unroll
            for (int r = 0; r < 8; ++r) {
                const int mm = tile * 16 + r + hi * 8;
                if (mm < NN) pre[(size_t)mm * 128 + t * 16 + row] = acc[t][r];
            }
    }
}

// Middle layers: pre[NN,128] = gather7(h)[NN,896] @ W (bf16 in, f32 out).
// Neighbor j constant across 4 K-chunks of 32 -> hoist index, contiguous lane loads.
__global__ __launch_bounds__(256) void gemm_mid(const unsigned short* __restrict__ h,
                                                const unsigned short* __restrict__ Wt,
                                                const int* __restrict__ idx,
                                                float* __restrict__ pre) {
    const int lane = threadIdx.x & 31;
    const int wave = threadIdx.x >> 5;
    const int tile = blockIdx.x * 8 + wave;
    if (tile >= MT) return;                      // uniform per wave
    const int row = lane & 15;
    const int hi  = lane >> 4;
    const int m   = tile * 16 + row;
    const int mc  = (m < NN) ? m : (NN - 1);
    const int alo = hi ? 8 : 0;
    const int blo = hi ? 8 : 0;

    int nodes[7];
#pragma unroll
    for (int j = 0; j < 7; ++j) nodes[j] = idx[mc * 7 + j];

    const v8f zacc = {0.f, 0.f, 0.f, 0.f, 0.f, 0.f, 0.f, 0.f};
    v8f acc[8];
#pragma unroll
    for (int t = 0; t < 8; ++t) acc[t] = zacc;

    for (int j = 0; j < 7; ++j) {
        const unsigned short* hrow = h + (size_t)nodes[j] * 128;
        if (j < 6) __builtin_prefetch(h + (size_t)nodes[j + 1] * 128, 0, 0);
#pragma unroll
        for (int q = 0; q < 4; ++q) {
            const int coff = q << 5;
            const unsigned short* ap = hrow + coff + alo;
            v16bf a = mk16(ap, ap + 16);
            const int kb = (j * 4 + q) * 32 + blo;
#pragma unroll
            for (int t = 0; t < 8; ++t) {
                const unsigned short* bp = Wt + (t * 16 + row) * 896 + kb;
                v16bf b = mk16(bp, bp + 16);
                acc[t] = __builtin_amdgcn_wmma_f32_16x16x32_bf16(false, a, false, b,
                                                                 (short)0, acc[t], false, false);
            }
        }
    }

    if (tile * 16 + 16 <= NN) {                  // uniform fast path
        float* pbase = pre + (size_t)(tile * 16 + hi * 8) * 128 + row;
#pragma unroll
        for (int t = 0; t < 8; ++t)
#pragma unroll
            for (int r = 0; r < 8; ++r)
                pbase[(size_t)r * 128 + t * 16] = acc[t][r];
    } else {
#pragma unroll
        for (int t = 0; t < 8; ++t)
#pragma unroll
            for (int r = 0; r < 8; ++r) {
                const int mm = tile * 16 + r + hi * 8;
                if (mm < NN) pre[(size_t)mm * 128 + t * 16 + row] = acc[t][r];
            }
    }
}

// Final layer: out[NN,36] = gather7(h) @ Wl + bl (3 N-tiles of 16, write n<36).
__global__ __launch_bounds__(256) void gemm_last(const unsigned short* __restrict__ h,
                                                 const unsigned short* __restrict__ Wlt,
                                                 const int* __restrict__ idx,
                                                 const float* __restrict__ bl,
                                                 float* __restrict__ out) {
    const int lane = threadIdx.x & 31;
    const int wave = threadIdx.x >> 5;
    const int tile = blockIdx.x * 8 + wave;
    if (tile >= MT) return;
    const int row = lane & 15;
    const int hi  = lane >> 4;
    const int m   = tile * 16 + row;
    const int mc  = (m < NN) ? m : (NN - 1);
    const int alo = hi ? 8 : 0;
    const int blo = hi ? 8 : 0;

    int nodes[7];
#pragma unroll
    for (int j = 0; j < 7; ++j) nodes[j] = idx[mc * 7 + j];

    const v8f zacc = {0.f, 0.f, 0.f, 0.f, 0.f, 0.f, 0.f, 0.f};
    v8f acc[3];
#pragma unroll
    for (int t = 0; t < 3; ++t) acc[t] = zacc;

    for (int j = 0; j < 7; ++j) {
        const unsigned short* hrow = h + (size_t)nodes[j] * 128;
        if (j < 6) __builtin_prefetch(h + (size_t)nodes[j + 1] * 128, 0, 0);
#pragma unroll
        for (int q = 0; q < 4; ++q) {
            const int coff = q << 5;
            const unsigned short* ap = hrow + coff + alo;
            v16bf a = mk16(ap, ap + 16);
            const int kb = (j * 4 + q) * 32 + blo;
#pragma unroll
            for (int t = 0; t < 3; ++t) {
                const unsigned short* bp = Wlt + (t * 16 + row) * 896 + kb;
                v16bf b = mk16(bp, bp + 16);
                acc[t] = __builtin_amdgcn_wmma_f32_16x16x32_bf16(false, a, false, b,
                                                                 (short)0, acc[t], false, false);
            }
        }
    }

    const bool full = (tile * 16 + 16 <= NN);    // uniform
#pragma unroll
    for (int t = 0; t < 3; ++t) {
        const int n = t * 16 + row;
        if (n < 36) {                            // divergent only for t==2
            const float bb = bl[n];
            if (full) {
#pragma unroll
                for (int r = 0; r < 8; ++r)
                    out[(size_t)(tile * 16 + r + hi * 8) * 36 + n] = acc[t][r] + bb;
            } else {
#pragma unroll
                for (int r = 0; r < 8; ++r) {
                    const int mm = tile * 16 + r + hi * 8;
                    if (mm < NN) out[(size_t)mm * 36 + n] = acc[t][r] + bb;
                }
            }
        }
    }
}

// ---------------- BN (training mode) ----------------
// Per-channel column sums: stats[c]=sum, stats[128+c]=sumsq.
__global__ __launch_bounds__(256) void colsum(const float* __restrict__ pre,
                                              float* __restrict__ stats) {
    __shared__ float s1[256], s2[256];
    const int c = blockIdx.x;         // 0..127
    float a = 0.f, b = 0.f;
    for (int v = threadIdx.x; v < NN; v += 256) {
        float t = pre[(size_t)v * 128 + c];
        a += t; b += t * t;
    }
    s1[threadIdx.x] = a; s2[threadIdx.x] = b;
    __syncthreads();
    for (int s = 128; s > 0; s >>= 1) {
        if (threadIdx.x < s) {
            s1[threadIdx.x] += s1[threadIdx.x + s];
            s2[threadIdx.x] += s2[threadIdx.x + s];
        }
        __syncthreads();
    }
    if (threadIdx.x == 0) { stats[c] = s1[0]; stats[128 + c] = s2[0]; }
}

// h = bf16(relu(g*(pre-mu)*rsqrt(var+eps)+be)). Bias of the linear cancels under BN.
__global__ void bnrelu(const float* __restrict__ pre, const float* __restrict__ stats,
                       const float* __restrict__ g, const float* __restrict__ be,
                       unsigned short* __restrict__ hout) {
    int i = blockIdx.x * blockDim.x + threadIdx.x;
    if (i >= NN * 128) return;
    int c = i & 127;
    const float inv = 1.0f / (float)NN;
    float mu  = stats[c] * inv;
    float var = stats[128 + c] * inv - mu * mu;
    float rs  = rsqrtf(var + 1e-5f);
    float v   = g[c] * (pre[i] - mu) * rs + be[c];
    hout[i] = f2bf(v > 0.f ? v : 0.f);
}

// ---------------- host launch ----------------
extern "C" void kernel_launch(void* const* d_in, const int* in_sizes, int n_in,
                              void* d_out, int out_size, void* d_ws, size_t ws_size,
                              hipStream_t stream) {
    const float* x   = (const float*)d_in[0];
    const int*   idx = (const int*)  d_in[1];
    const float* W0  = (const float*)d_in[2];
    // d_in[3] = b0 : cancels under training-mode BN, unused
    const float* g0  = (const float*)d_in[4];
    const float* be0 = (const float*)d_in[5];
    const float* Wm  = (const float*)d_in[6];
    // d_in[7] = bm : cancels under training-mode BN, unused
    const float* gm  = (const float*)d_in[8];
    const float* bem = (const float*)d_in[9];
    const float* Wl  = (const float*)d_in[10];
    const float* bl  = (const float*)d_in[11];
    float* out = (float*)d_out;

    // workspace layout (256B aligned)
    char* ws = (char*)d_ws;
    size_t o = 0;
    auto take = [&](size_t bytes) { char* p = ws + o; o = (o + bytes + 255) & ~(size_t)255; return p; };
    float*          pre   = (float*)         take((size_t)NN * 128 * 4);
    unsigned short* hA    = (unsigned short*)take((size_t)NN * 128 * 2);
    unsigned short* hB    = (unsigned short*)take((size_t)NN * 128 * 2);
    unsigned short* A0    = (unsigned short*)take((size_t)NN * 32 * 2);
    unsigned short* W0t   = (unsigned short*)take((size_t)128 * 32 * 2);
    unsigned short* Wmt   = (unsigned short*)take((size_t)14 * 128 * 896 * 2);
    unsigned short* Wlt   = (unsigned short*)take((size_t)48 * 896 * 2);
    float*          stats = (float*)         take(256 * 4);
    (void)ws_size; (void)in_sizes; (void)n_in; (void)out_size;

    // prep: bf16 weights (n-major) + gathered/padded layer-0 input
    k_w0t<<<(128 * 32 + 255) / 256, 256, 0, stream>>>(W0, W0t);
    k_wmt<<<(14 * 128 * 896 + 255) / 256, 256, 0, stream>>>(Wm, Wmt);
    k_wlt<<<(48 * 896 + 255) / 256, 256, 0, stream>>>(Wl, Wlt);
    k_a0 <<<(NN * 32 + 255) / 256, 256, 0, stream>>>(x, idx, A0);

    const int elemBlocks = (NN * 128 + 255) / 256;

    // layer 0
    gemm0<<<GBLK, 256, 0, stream>>>(A0, W0t, pre);
    colsum<<<128, 256, 0, stream>>>(pre, stats);
    bnrelu<<<elemBlocks, 256, 0, stream>>>(pre, stats, g0, be0, hA);

    // 14 middle layers (ping-pong hA/hB)
    unsigned short* hcur = hA;
    unsigned short* hnxt = hB;
    for (int l = 0; l < 14; ++l) {
        gemm_mid<<<GBLK, 256, 0, stream>>>(hcur, Wmt + (size_t)l * 128 * 896, idx, pre);
        colsum<<<128, 256, 0, stream>>>(pre, stats);
        bnrelu<<<elemBlocks, 256, 0, stream>>>(pre, stats, gm + l * 128, bem + l * 128, hnxt);
        unsigned short* t = hcur; hcur = hnxt; hnxt = t;
    }

    // final classifier layer (with bias, fp32 out)
    gemm_last<<<GBLK, 256, 0, stream>>>(hcur, Wlt, idx, bl, out);
}
